// GatedCrossAttention_56014963475219
// MI455X (gfx1250) — matrix operand order repaired and checked by reading, not compiled, not run on hardware
//
#include <hip/hip_runtime.h>
#include <hip/hip_bf16.h>
#include <stdint.h>

// GatedCrossAttention for MI455X (gfx1250): bf16 WMMA pipeline.
// B=32768, L=3, D=1024, H=16, DH=64.

#define DCFG 1024
#define HCFG 16
#define NBATCH 32768
#define LN_EPS 1e-5f
#define SCHED_FENCE() __builtin_amdgcn_sched_barrier(0)

typedef __bf16    v16bf  __attribute__((ext_vector_type(16)));
typedef __bf16    bf16x2 __attribute__((ext_vector_type(2)));
typedef float     v8f    __attribute__((ext_vector_type(8)));
typedef float     f32x2  __attribute__((ext_vector_type(2)));
typedef uint32_t  u32x8  __attribute__((ext_vector_type(8)));
typedef uint32_t  u32x4  __attribute__((ext_vector_type(4)));
typedef uint32_t  u32x2  __attribute__((ext_vector_type(2)));
typedef float     f32x4  __attribute__((ext_vector_type(4)));

// Packed f32x2 -> bf16x2 (RNE): single v_cvt_pk_bf16_f32.
__device__ __forceinline__ uint32_t pk2(float a, float b) {
  f32x2 t; t[0] = a; t[1] = b;
  bf16x2 h = __builtin_convertvector(t, bf16x2);
  return __builtin_bit_cast(uint32_t, h);
}
__device__ __forceinline__ uint16_t f2bf(float f) {
  return __builtin_bit_cast(uint16_t, (__bf16)f);
}
__device__ __forceinline__ float bf_lo(uint32_t w) { return __uint_as_float(w << 16); }
__device__ __forceinline__ float bf_hi(uint32_t w) { return __uint_as_float(w & 0xffff0000u); }

__device__ __forceinline__ v8f wmma_bf16(u32x8 a, u32x8 b, v8f c) {
  return __builtin_amdgcn_wmma_f32_16x16x32_bf16(
      false, __builtin_bit_cast(v16bf, a),
      false, __builtin_bit_cast(v16bf, b),
      (short)0, c, false, false);
}

// A fragment at per-lane pointer p (row + 16*lg already applied): 16
// contiguous K elements at p and p+8 (works for LDS or global).
__device__ __forceinline__ u32x8 afrag_at(const uint16_t* p) {
  u32x4 lo = *(const u32x4*)p;
  u32x4 hi = *(const u32x4*)(p + 8);
  u32x8 r;
  r[0] = lo[0]; r[1] = lo[1]; r[2] = lo[2]; r[3] = lo[3];
  r[4] = hi[0]; r[5] = hi[1]; r[6] = hi[2]; r[7] = hi[3];
  return r;
}
// B fragment at per-lane pointer (n-row + 16*lg already applied): 16
// contiguous K elements (32B).
__device__ __forceinline__ u32x8 bfrag_at(const uint16_t* p) {
  return *(const u32x8*)p;
}

// ---------------- Kernel 0: f32 -> bf16 conversion -----------------------
__global__ void cvt_f32_bf16(const float* __restrict__ src,
                             uint16_t* __restrict__ dst, int n4) {
  int i = blockIdx.x * blockDim.x + threadIdx.x;
  if (i >= n4) return;
  f32x4 v = *(const f32x4*)(src + 4 * (size_t)i);
  u32x2 o; o[0] = pk2(v[0], v[1]); o[1] = pk2(v[2], v[3]);
  *(u32x2*)(dst + 4 * (size_t)i) = o;
}

// ---------------- Kernel 1: Q = q @ Wq^T  (bf16 in/out) ------------------
// Workgroup 256 thr = 8 waves, tile M=64 x N=128; wave = 16 rows x 64 cols.
// Pointer-carried, 2-phase ping-pong k-loop; sched fences pin interleave.
// launch_bounds(256,2): 2 waves/EU occupancy floor -> ~2x VGPR budget,
// room for the double-buffered working set without scratch spills.
__global__ void __launch_bounds__(256, 2)
qproj_kernel(const uint16_t* __restrict__ qbf,
             const uint16_t* __restrict__ Wq,
             uint16_t* __restrict__ Qbf) {
  const int lane = threadIdx.x & 31, wv = threadIdx.x >> 5;
  const int lm = lane & 15, lg = lane >> 4;
  const int my = wv & 3, nx = wv >> 2;
  const int row0 = blockIdx.y * 64 + my * 16;
  const int n0   = blockIdx.x * 128 + nx * 64;
  v8f acc[4] = {};

  const uint16_t* ap = qbf + (row0 + lm) * DCFG + 16 * lg;
  const uint16_t* bp = Wq + (n0 + lm) * DCFG + 16 * lg;

  u32x8 aA, aB, bA[4], bB[4];
  aA = afrag_at(ap);
#pragma unroll
  for (int i = 0; i < 4; ++i) bA[i] = bfrag_at(bp + i * 16 * DCFG);

  for (int it = 0; it < 15; ++it) {
    aB = afrag_at(ap + 32);
#pragma unroll
    for (int i = 0; i < 4; ++i) bB[i] = bfrag_at(bp + i * 16 * DCFG + 32);
    SCHED_FENCE();
#pragma unroll
    for (int i = 0; i < 4; ++i) acc[i] = wmma_bf16(aA, bA[i], acc[i]);
    SCHED_FENCE();
    ap += 64; bp += 64;
    aA = afrag_at(ap);
#pragma unroll
    for (int i = 0; i < 4; ++i) bA[i] = bfrag_at(bp + i * 16 * DCFG);
    SCHED_FENCE();
#pragma unroll
    for (int i = 0; i < 4; ++i) acc[i] = wmma_bf16(aB, bB[i], acc[i]);
    SCHED_FENCE();
  }
  // epilogue: A/B sets hold k = 960; load 992 and finish
  aB = afrag_at(ap + 32);
#pragma unroll
  for (int i = 0; i < 4; ++i) bB[i] = bfrag_at(bp + i * 16 * DCFG + 32);
#pragma unroll
  for (int i = 0; i < 4; ++i) acc[i] = wmma_bf16(aA, bA[i], acc[i]);
#pragma unroll
  for (int i = 0; i < 4; ++i) acc[i] = wmma_bf16(aB, bB[i], acc[i]);

#pragma unroll
  for (int i = 0; i < 4; ++i)
#pragma unroll
    for (int r = 0; r < 8; ++r) {
      int m = row0 + r + 8 * lg;
      int n = n0 + 16 * i + lm;
      Qbf[m * DCFG + n] = f2bf(acc[i][r]);
    }
}

// ------ Kernel 2: fused K/V/R projection + gated attention ---------------
// One workgroup = 16 batches (48 x-rows). x staged in LDS as bf16 (96KB).
// Loop over 8 head-pairs: stacked GEMM [48 x 384] (K|V|R of 2 heads),
// 8 waves x 3 ntiles x 3 mtiles. Global B streams pointer-carried and
// ping-pong double-buffered; LDS A fragments loaded per-step.
__global__ void __launch_bounds__(256, 2)
kvr_attn_kernel(const float* __restrict__ x,
                const uint16_t* __restrict__ Wk,
                const uint16_t* __restrict__ Wv,
                const uint16_t* __restrict__ Wr,
                const uint16_t* __restrict__ Qbf,
                uint16_t* __restrict__ AObf,
                float* __restrict__ attnOut) {
  extern __shared__ char smem[];
  uint16_t* sX   = (uint16_t*)smem;                              // 48*1024 bf16
  uint16_t* sKVR = (uint16_t*)(smem + 48 * 1024 * 2);            // 48*384 bf16
  float*    sS   = (float*)(smem + 48 * 1024 * 2 + 48 * 384 * 2);// 16*2*3 f32
  const int tid = threadIdx.x;
  const int lane = tid & 31, wv = tid >> 5;
  const int lm = lane & 15, lg = lane >> 4;
  const int b0 = blockIdx.x * 16;

  { // stage 48 rows of x, converting f32 -> bf16
    const float* src = x + (size_t)b0 * 3 * DCFG;
    for (int it = 0; it < 48; ++it) {
      int i4 = tid + it * 256;                 // 12288 float4 total
      f32x4 v = *(const f32x4*)(src + 4 * (size_t)i4);
      u32x2 o; o[0] = pk2(v[0], v[1]); o[1] = pk2(v[2], v[3]);
      *(u32x2*)(sX + 4 * (size_t)i4) = o;
    }
  }
  __syncthreads();

  const uint16_t* Wmats[3] = {Wk, Wv, Wr};
  const int nt0 = wv * 3;                       // this wave's 3 ntiles of 24
  const uint16_t* xp0 = sX + lm * DCFG + 16 * lg;

  for (int hp = 0; hp < 8; ++hp) {
    v8f acc[3][3] = {};                         // [ntile][mtile]
    const uint16_t* wp[3];                      // per-lane weight streams
#pragma unroll
    for (int i = 0; i < 3; ++i) {
      int nt = nt0 + i;
      int ho = nt / 12, rem = nt % 12, mat = rem >> 2, cq = rem & 3;
      int h = hp * 2 + ho;
      wp[i] = Wmats[mat] + (h * 64 + cq * 16 + lm) * DCFG + 16 * lg;
    }
    const uint16_t* xp = xp0;

    u32x8 bA[3], bB[3], a[3];
#pragma unroll
    for (int i = 0; i < 3; ++i) bA[i] = bfrag_at(wp[i]);

    for (int it = 0; it < 15; ++it) {
#pragma unroll
      for (int i = 0; i < 3; ++i) bB[i] = bfrag_at(wp[i] + 32);
#pragma unroll
      for (int mt = 0; mt < 3; ++mt) a[mt] = afrag_at(xp + mt * 16 * DCFG);
      SCHED_FENCE();
#pragma unroll
      for (int i = 0; i < 3; ++i)
#pragma unroll
        for (int mt = 0; mt < 3; ++mt) acc[i][mt] = wmma_bf16(a[mt], bA[i], acc[i][mt]);
      SCHED_FENCE();
#pragma unroll
      for (int i = 0; i < 3; ++i) { wp[i] += 64; bA[i] = bfrag_at(wp[i]); }
#pragma unroll
      for (int mt = 0; mt < 3; ++mt) a[mt] = afrag_at(xp + mt * 16 * DCFG + 32);
      SCHED_FENCE();
#pragma unroll
      for (int i = 0; i < 3; ++i)
#pragma unroll
        for (int mt = 0; mt < 3; ++mt) acc[i][mt] = wmma_bf16(a[mt], bB[i], acc[i][mt]);
      SCHED_FENCE();
      xp += 64;
    }
    // epilogue: bA holds k = 960; last two steps
#pragma unroll
    for (int i = 0; i < 3; ++i) bB[i] = bfrag_at(wp[i] + 32);
#pragma unroll
    for (int mt = 0; mt < 3; ++mt) a[mt] = afrag_at(xp + mt * 16 * DCFG);
#pragma unroll
    for (int i = 0; i < 3; ++i)
#pragma unroll
      for (int mt = 0; mt < 3; ++mt) acc[i][mt] = wmma_bf16(a[mt], bA[i], acc[i][mt]);
#pragma unroll
    for (int mt = 0; mt < 3; ++mt) a[mt] = afrag_at(xp + mt * 16 * DCFG + 32);
#pragma unroll
    for (int i = 0; i < 3; ++i)
#pragma unroll
      for (int mt = 0; mt < 3; ++mt) acc[i][mt] = wmma_bf16(a[mt], bB[i], acc[i][mt]);

    // dump stacked result to LDS: col = ho*192 + mat*64 + d  (== nt*16+lm)
#pragma unroll
    for (int i = 0; i < 3; ++i) {
      int nt = nt0 + i;
#pragma unroll
      for (int mt = 0; mt < 3; ++mt)
#pragma unroll
        for (int r = 0; r < 8; ++r) {
          int row = mt * 16 + r + 8 * lg;
          int col = nt * 16 + lm;
          sKVR[row * 384 + col] = f2bf(acc[i][mt][r]);
        }
    }
    __syncthreads();

    // scores: 96 tasks = 16 batches * 2 heads * 3 keys (vectorized b128 reads)
    if (tid < 96) {
      int bi = tid & 15, rest = tid >> 4;       // rest in 0..5
      int ho = rest / 3, l = rest % 3;
      int h = hp * 2 + ho;
      const uint16_t* qp = Qbf + (b0 + bi) * DCFG + h * 64;
      const uint16_t* kp = sKVR + (bi * 3 + l) * 384 + ho * 192;
      float s = 0.f;
#pragma unroll
      for (int c = 0; c < 8; ++c) {
        u32x4 qv = *(const u32x4*)(qp + c * 8);
        u32x4 kv = *(const u32x4*)(kp + c * 8);
#pragma unroll
        for (int j = 0; j < 4; ++j)
          s += bf_lo(qv[j]) * bf_lo(kv[j]) + bf_hi(qv[j]) * bf_hi(kv[j]);
      }
      sS[(bi * 2 + ho) * 3 + l] = s * 0.125f;   // 1/sqrt(64)
    }
    __syncthreads();

    // softmax over L=3 + write attn probs
    if (tid < 32) {
      int bi = tid & 15, ho = tid >> 4;
      int h = hp * 2 + ho;
      float* sp = sS + (bi * 2 + ho) * 3;
      float m = fmaxf(sp[0], fmaxf(sp[1], sp[2]));
      float e0 = __expf(sp[0] - m), e1 = __expf(sp[1] - m), e2 = __expf(sp[2] - m);
      float inv = 1.f / (e0 + e1 + e2);
      float p0 = e0 * inv, p1 = e1 * inv, p2 = e2 * inv;
      sp[0] = p0; sp[1] = p1; sp[2] = p2;
      float* ap2 = attnOut + ((size_t)(b0 + bi) * HCFG + h) * 3;
      ap2[0] = p0; ap2[1] = p1; ap2[2] = p2;
    }
    __syncthreads();

    // gated combine: out[b,h,d] = sum_l p_l * sigmoid(R)*V ; 2048 vals, 8/thread
    {
      int base = tid * 8;
      int bi = base >> 7, ho = (base >> 6) & 1, d0 = base & 63;
      int h = hp * 2 + ho;
      const float* pp = sS + (bi * 2 + ho) * 3;
      float o[8] = {};
#pragma unroll
      for (int l = 0; l < 3; ++l) {
        float pl = pp[l];
        const uint16_t* rowp = sKVR + (bi * 3 + l) * 384 + ho * 192;
        u32x4 vv = *(const u32x4*)(rowp + 64 + d0);
        u32x4 rr = *(const u32x4*)(rowp + 128 + d0);
#pragma unroll
        for (int j = 0; j < 4; ++j) {
          float v0 = bf_lo(vv[j]), v1 = bf_hi(vv[j]);
          float r0 = bf_lo(rr[j]), r1 = bf_hi(rr[j]);
          o[2 * j]     += pl * v0 * (1.f / (1.f + __expf(-r0)));
          o[2 * j + 1] += pl * v1 * (1.f / (1.f + __expf(-r1)));
        }
      }
      u32x4 pkd;
#pragma unroll
      for (int j = 0; j < 4; ++j) pkd[j] = pk2(o[2 * j], o[2 * j + 1]);
      *(u32x4*)(AObf + (b0 + bi) * DCFG + h * 64 + d0) = pkd;
    }
    __syncthreads();
  }
}

// ---- Kernel 3: O = AO @ Wo^T, then LayerNorm(q + O) fused ---------------
// Workgroup = 16 full rows (N=1024). 8 waves; each wave does two 64-col
// half-passes (4 ntiles each) with the pointer-carried pipelined k-loop.
__global__ void __launch_bounds__(256, 2)
oproj_ln_kernel(const uint16_t* __restrict__ AObf,
                const uint16_t* __restrict__ Wo,
                const float* __restrict__ q,
                const float* __restrict__ gamma,
                const float* __restrict__ beta,
                float* __restrict__ out) {
  extern __shared__ char smem[];
  uint16_t* sA   = (uint16_t*)smem;                         // 16*1024 bf16
  float*    sY   = (float*)(smem + 16 * 1024 * 2);          // 16*1024 f32
  float*    sRed = (float*)(smem + 16 * 1024 * 2 + 16 * 1024 * 4); // 16*16*2
  const int tid = threadIdx.x;
  const int lane = tid & 31, wv = tid >> 5;
  const int lm = lane & 15, lg = lane >> 4;
  const int row0 = blockIdx.x * 16;

  { // stage 16 AO rows (bf16 copy)
    const u32x4* src = (const u32x4*)(AObf + (size_t)row0 * DCFG);
    u32x4* dst = (u32x4*)sA;
    for (int it = 0; it < 8; ++it) dst[tid + it * 256] = src[tid + it * 256];
  }
  __syncthreads();

  for (int half = 0; half < 2; ++half) {
    const int n0 = wv * 128 + half * 64;
    v8f acc[4] = {};
    const uint16_t* ap = sA + lm * DCFG + 16 * lg;
    const uint16_t* bp = Wo + (n0 + lm) * DCFG + 16 * lg;
    u32x8 aA, aB, bA[4], bB[4];
    aA = afrag_at(ap);
#pragma unroll
    for (int i = 0; i < 4; ++i) bA[i] = bfrag_at(bp + i * 16 * DCFG);

    for (int it = 0; it < 15; ++it) {
      aB = afrag_at(ap + 32);
#pragma unroll
      for (int i = 0; i < 4; ++i) bB[i] = bfrag_at(bp + i * 16 * DCFG + 32);
      SCHED_FENCE();
#pragma unroll
      for (int i = 0; i < 4; ++i) acc[i] = wmma_bf16(aA, bA[i], acc[i]);
      SCHED_FENCE();
      ap += 64; bp += 64;
      aA = afrag_at(ap);
#pragma unroll
      for (int i = 0; i < 4; ++i) bA[i] = bfrag_at(bp + i * 16 * DCFG);
      SCHED_FENCE();
#pragma unroll
      for (int i = 0; i < 4; ++i) acc[i] = wmma_bf16(aB, bB[i], acc[i]);
      SCHED_FENCE();
    }
    aB = afrag_at(ap + 32);
#pragma unroll
    for (int i = 0; i < 4; ++i) bB[i] = bfrag_at(bp + i * 16 * DCFG + 32);
#pragma unroll
    for (int i = 0; i < 4; ++i) acc[i] = wmma_bf16(aA, bA[i], acc[i]);
#pragma unroll
    for (int i = 0; i < 4; ++i) acc[i] = wmma_bf16(aB, bB[i], acc[i]);

#pragma unroll
    for (int i = 0; i < 4; ++i)
#pragma unroll
      for (int r = 0; r < 8; ++r)
        sY[(r + 8 * lg) * DCFG + n0 + 16 * i + lm] = acc[i][r];
  }
  __syncthreads();

  // LayerNorm: 16 threads per row, 64 contiguous cols each
  int r = tid >> 4, cg = tid & 15;
  int c0 = cg * 64;
  const float* qrow = q + (size_t)(row0 + r) * DCFG;
  float yv[64];
  float s = 0.f, s2 = 0.f;
#pragma unroll
  for (int j = 0; j < 64; ++j) {
    float v = sY[r * DCFG + c0 + j] + qrow[c0 + j];
    yv[j] = v; s += v; s2 += v * v;
  }
  sRed[(r * 16 + cg) * 2]     = s;
  sRed[(r * 16 + cg) * 2 + 1] = s2;
  __syncthreads();
  float ts = 0.f, ts2 = 0.f;
#pragma unroll
  for (int i = 0; i < 16; ++i) {
    ts  += sRed[(r * 16 + i) * 2];
    ts2 += sRed[(r * 16 + i) * 2 + 1];
  }
  float mu  = ts * (1.f / 1024.f);
  float var = ts2 * (1.f / 1024.f) - mu * mu;
  float rstd = rsqrtf(var + LN_EPS);
  float* orow = out + (size_t)(row0 + r) * DCFG;
#pragma unroll
  for (int j = 0; j < 64; ++j)
    orow[c0 + j] = (yv[j] - mu) * rstd * gamma[c0 + j] + beta[c0 + j];
}

// ------------------------------ launch -----------------------------------
extern "C" void kernel_launch(void* const* d_in, const int* in_sizes, int n_in,
                              void* d_out, int out_size, void* d_ws, size_t ws_size,
                              hipStream_t stream) {
  const float* q  = (const float*)d_in[0];
  const float* x  = (const float*)d_in[1];
  const float* Wq = (const float*)d_in[2];
  const float* Wk = (const float*)d_in[3];
  const float* Wv = (const float*)d_in[4];
  const float* Wr = (const float*)d_in[5];
  const float* Wo = (const float*)d_in[6];
  const float* gamma = (const float*)d_in[7];
  const float* beta  = (const float*)d_in[8];

  float* out  = (float*)d_out;                       // [B*D]
  float* attn = out + (size_t)NBATCH * DCFG;         // [B*H*3]

  const size_t DD = (size_t)DCFG * DCFG;
  const size_t BD = (size_t)NBATCH * DCFG;
  uint16_t* wq_bf  = (uint16_t*)d_ws;
  uint16_t* wk_bf  = wq_bf + DD;
  uint16_t* wv_bf  = wq_bf + 2 * DD;
  uint16_t* wr_bf  = wq_bf + 3 * DD;
  uint16_t* wo_bf  = wq_bf + 4 * DD;
  uint16_t* qbf_in = wq_bf + 5 * DD;                 // q converted, [B*D] bf16
  uint16_t* Qbf    = qbf_in + BD;                    // projected Q, [B*D] bf16
  uint16_t* AObf   = Qbf + BD;                       // attention out, [B*D] bf16

  const int n4w = (int)(DD / 4);
  dim3 cblk(256), cgrdw((n4w + 255) / 256);
  cvt_f32_bf16<<<cgrdw, cblk, 0, stream>>>(Wq, wq_bf, n4w);
  cvt_f32_bf16<<<cgrdw, cblk, 0, stream>>>(Wk, wk_bf, n4w);
  cvt_f32_bf16<<<cgrdw, cblk, 0, stream>>>(Wv, wv_bf, n4w);
  cvt_f32_bf16<<<cgrdw, cblk, 0, stream>>>(Wr, wr_bf, n4w);
  cvt_f32_bf16<<<cgrdw, cblk, 0, stream>>>(Wo, wo_bf, n4w);
  const int n4q = (int)(BD / 4);
  cvt_f32_bf16<<<(n4q + 255) / 256, cblk, 0, stream>>>(q, qbf_in, n4q);

  qproj_kernel<<<dim3(DCFG / 128, NBATCH / 64), 256, 0, stream>>>(qbf_in, wq_bf, Qbf);

  size_t lds2 = 48 * 1024 * 2 + 48 * 384 * 2 + 16 * 2 * 3 * 4;   // ~135 KB
  kvr_attn_kernel<<<NBATCH / 16, 256, lds2, stream>>>(
      x, wk_bf, wv_bf, wr_bf, Qbf, AObf, attn);

  size_t lds3 = 16 * 1024 * 2 + 16 * 1024 * 4 + 16 * 16 * 2 * 4; // ~98 KB
  oproj_ln_kernel<<<NBATCH / 16, 256, lds3, stream>>>(
      AObf, wo_bf, q, gamma, beta, out);
}
